// GNN_8383776162106
// MI455X (gfx1250) — matrix operand order repaired and checked by reading, not compiled, not run on hardware
//
#include <hip/hip_runtime.h>

#define N_NODES 100000
#define N_EDGES 3200000
#define DIM     256
#define NODE_BYTES ((size_t)N_NODES * DIM * sizeof(float))   // 102,400,000

typedef __attribute__((ext_vector_type(16))) __bf16 v16bf;
typedef __attribute__((ext_vector_type(8)))  float  v8f;

// ---------------- degree / normalization ----------------

__global__ void gcn_deg_init(float* __restrict__ deg) {
    int i = blockIdx.x * 256 + threadIdx.x;
    if (i < N_NODES) deg[i] = 1.0f;                 // self-loop contributes 1
}

__global__ void gcn_deg_edges(const int* __restrict__ dst, float* __restrict__ deg) {
    int i = blockIdx.x * 256 + threadIdx.x;
    if (i < N_EDGES) unsafeAtomicAdd(&deg[dst[i]], 1.0f);   // global_atomic_add_f32
}

__global__ void gcn_deg_finalize(float* __restrict__ deg) {
    int i = blockIdx.x * 256 + threadIdx.x;
    if (i < N_NODES) deg[i] = rsqrtf(deg[i]);       // deg >= 1 always
}

// ---------------- W fp32 -> bf16, pre-swizzled into WMMA B-matrix layout ----
// For K-slab kt (32 K values) and N-tile nt (16 cols), lane L holds column
// L%16, and its 16 bf16 elements are K = (L/16)*16 + e  (e = 0..15).
// Stored contiguously: Wb[(((kt*16)+nt)*32 + lane)*16 + e]  -> one 32B load.

__global__ void gcn_convW(const float* __restrict__ W, __bf16* __restrict__ Wb) {
    int tid   = blockIdx.x * 256 + threadIdx.x;     // 0..4095
    int kt    = tid >> 9;                           // 0..7
    int rem   = tid & 511;
    int nt    = rem >> 5;                           // 0..15
    int lane  = rem & 31;
    int half  = lane >> 4;
    int n     = nt * 16 + (lane & 15);
    __bf16* out = Wb + ((size_t)((kt * 16 + nt) * 32 + lane)) * 16;
#pragma unroll
    for (int e = 0; e < 16; ++e) {
        int k = kt * 32 + half * 16 + e;
        out[e] = (__bf16)W[k * DIM + n];
    }
}

// ---------------- GEMM: H = X @ W via v_wmma_f32_16x16x32_bf16 -------------
// Block = 256 threads = 8 waves. Wave computes a 16-row x 64-col strip:
// 2 M-tiles per block (waves 0-3 / 4-7), 4 N-subtiles of 16 per wave.
// K = 256 processed as 8 slabs of 32.

__global__ __launch_bounds__(256)
void gcn_gemm_wmma(const float* __restrict__ X, const __bf16* __restrict__ Wb,
                   float* __restrict__ H) {
    const int lane  = threadIdx.x & 31;
    const int wave  = threadIdx.x >> 5;
    const int mtile = blockIdx.x * 2 + (wave >> 2);
    const int nq    = wave & 3;                     // 64-col quadrant
    const int half  = lane >> 4;
    const int mrow  = mtile * 16 + (lane & 15);
    const float* xrow = X + (size_t)mrow * DIM;

    v8f acc[4];
    acc[0] = (v8f)0.0f; acc[1] = (v8f)0.0f; acc[2] = (v8f)0.0f; acc[3] = (v8f)0.0f;

    const v16bf* wb = (const v16bf*)Wb;

#pragma unroll
    for (int kc = 0; kc < 8; ++kc) {
        // A tile (16x32 bf16): elems 0-15 = K half*16..+15
        const float* p0 = xrow + kc * 32 + half * 16;
        float4 f0 = *(const float4*)(p0);
        float4 f1 = *(const float4*)(p0 + 4);
        float4 f2 = *(const float4*)(p0 + 8);
        float4 f3 = *(const float4*)(p0 + 12);
        union { v16bf v; __bf16 e[16]; } A;
        A.e[0]  = (__bf16)f0.x; A.e[1]  = (__bf16)f0.y;
        A.e[2]  = (__bf16)f0.z; A.e[3]  = (__bf16)f0.w;
        A.e[4]  = (__bf16)f1.x; A.e[5]  = (__bf16)f1.y;
        A.e[6]  = (__bf16)f1.z; A.e[7]  = (__bf16)f1.w;
        A.e[8]  = (__bf16)f2.x; A.e[9]  = (__bf16)f2.y;
        A.e[10] = (__bf16)f2.z; A.e[11] = (__bf16)f2.w;
        A.e[12] = (__bf16)f3.x; A.e[13] = (__bf16)f3.y;
        A.e[14] = (__bf16)f3.z; A.e[15] = (__bf16)f3.w;

#pragma unroll
        for (int nt = 0; nt < 4; ++nt) {
            v16bf b = wb[(size_t)((kc * 16 + nq * 4 + nt) * 32 + lane)];
            acc[nt] = __builtin_amdgcn_wmma_f32_16x16x32_bf16(
                false, A.v, false, b, (short)0, acc[nt], false, false);
        }
    }

    // D layout: VGPR v holds row (v + 8*half), col = lane%16 within tile
#pragma unroll
    for (int nt = 0; nt < 4; ++nt) {
#pragma unroll
        for (int v = 0; v < 8; ++v) {
            size_t r = (size_t)(mtile * 16 + half * 8 + v);
            H[r * DIM + nq * 64 + nt * 16 + (lane & 15)] = acc[nt][v];
        }
    }
}

// ---------------- aggregation -------------------------------------------
// out[i] = b + dis[i]^2 * h[i]  (self-loop term), then edge scatter-adds.

__global__ void gcn_agg_init(const float* __restrict__ h, const float* __restrict__ bias,
                             const float* __restrict__ dis, float* __restrict__ out) {
    int n = blockIdx.x;                 // one node per block, 256 = DIM threads
    int c = threadIdx.x;
    float dn = dis[n];
    size_t idx = (size_t)n * DIM + c;
    out[idx] = bias[c] + h[idx] * dn * dn;
}

__global__ __launch_bounds__(256)
void gcn_agg_edges(const int* __restrict__ src, const int* __restrict__ dst,
                   const float* __restrict__ h, const float* __restrict__ dis,
                   float* __restrict__ out) {
    int e = blockIdx.x * 8 + (threadIdx.x >> 5);    // one wave32 per edge
    if (e >= N_EDGES) return;
    int lane = threadIdx.x & 31;
    int s = src[e];
    int d = dst[e];
    float w = dis[s] * dis[d];
    const float* hr = h + (size_t)s * DIM;
    float* orow = out + (size_t)d * DIM;
#pragma unroll
    for (int j = 0; j < 8; ++j) {
        int c = lane + j * 32;
        unsafeAtomicAdd(&orow[c], hr[c] * w);       // L2-resident fp32 atomic
    }
}

// ---------------- launch ----------------------------------------------------

extern "C" void kernel_launch(void* const* d_in, const int* in_sizes, int n_in,
                              void* d_out, int out_size, void* d_ws, size_t ws_size,
                              hipStream_t stream) {
    const float* x   = (const float*)d_in[0];
    const int*   ei  = (const int*)d_in[1];
    const int*   src = ei;
    const int*   dst = ei + N_EDGES;
    const float* W[3] = { (const float*)d_in[2], (const float*)d_in[4], (const float*)d_in[6] };
    const float* B[3] = { (const float*)d_in[3], (const float*)d_in[5], (const float*)d_in[7] };

    char* ws = (char*)d_ws;
    float*  dis  = (float*)ws;                                  // N floats
    __bf16* Wb   = (__bf16*)(ws + (512u << 10));                // 128 KB
    float*  hbuf = (float*)(ws + (1u << 20));                   // N*D floats
    float*  nbuf = (float*)(ws + (1u << 20) + NODE_BYTES);      // N*D floats
    float*  outp = (float*)d_out;

    const int NB256 = (N_NODES + 255) / 256;       // 391
    const int EB256 = (N_EDGES + 255) / 256;       // 12500
    const int GEMMB = N_NODES / 32;                // 3125 (100000 % 32 == 0)
    const int EDGEB = N_EDGES / 8;                 // 400000

    // symmetric normalization (computed once, reused by all layers)
    gcn_deg_init<<<NB256, 256, 0, stream>>>(dis);
    gcn_deg_edges<<<EB256, 256, 0, stream>>>(dst, dis);
    gcn_deg_finalize<<<NB256, 256, 0, stream>>>(dis);

    const float* xin = x;
    for (int l = 0; l < 3; ++l) {
        float* xout = (l == 2) ? outp : nbuf;
        gcn_convW<<<16, 256, 0, stream>>>(W[l], Wb);
        gcn_gemm_wmma<<<GEMMB, 256, 0, stream>>>(xin, Wb, hbuf);
        gcn_agg_init<<<N_NODES, 256, 0, stream>>>(hbuf, B[l], dis, xout);
        gcn_agg_edges<<<EDGEB, 256, 0, stream>>>(src, dst, hbuf, dis, xout);
        xin = xout;
    }
}
